// _SelfAttentionBlock_4861902979800
// MI455X (gfx1250) — compile-verified, matrix operand order checked by reading, not run on hardware
//
#include <hip/hip_runtime.h>
#include <hip/hip_bf16.h>

// ---------------- problem constants ----------------
static constexpr int Bn  = 2;       // batch
static constexpr int C   = 512;     // input channels
static constexpr int N   = 6400;    // H*W = 80*80
static constexpr int CK  = 256;     // key/query channels
static constexpr int CV  = 256;     // value channels
static constexpr int CO  = 512;     // output channels

typedef __attribute__((ext_vector_type(16))) __bf16 v16bf;
typedef __attribute__((ext_vector_type(8)))  float  v8f;

// LDS layout for the flash kernel (units: bf16 elements)
static constexpr int KT_STRIDE = 264;             // 256 + 8 pad (528B, 16B mult)
static constexpr int VT_STRIDE = 40;              // 32 + 8 pad  (80B, 16B mult)
static constexpr int KT_ELEMS  = 32 * KT_STRIDE;  // 8448 per buffer
static constexpr int VT_ELEMS  = 256 * VT_STRIDE; // 10240 per buffer
static constexpr int P_STRIDE  = 40;
static constexpr int P_ELEMS   = 16 * P_STRIDE;   // 640 per wave
static constexpr int O_STRIDE  = 264;
static constexpr int SM_VT0    = 2 * KT_ELEMS;            // after 2 K buffers
static constexpr int SM_PW0    = SM_VT0 + 2 * VT_ELEMS;   // after 2 V buffers
static constexpr int SM_TOTAL  = SM_PW0 + 4 * P_ELEMS;    // 39936 elems
static constexpr int SM_BYTES  = SM_TOTAL * 2;            // 79872 B

__device__ __forceinline__ v8f v8zero() {
  v8f z = {0.f, 0.f, 0.f, 0.f, 0.f, 0.f, 0.f, 0.f};
  return z;
}

__device__ __forceinline__ v8f wmma_bf16(v16bf a, v16bf b, v8f c) {
  // D = A(16x32 bf16) * B(32x16 bf16) + C(16x16 f32)
  return __builtin_amdgcn_wmma_f32_16x16x32_bf16(
      /*neg_a=*/false, a, /*neg_b=*/false, b,
      /*c_mod=*/(short)0, c, /*reuse_a=*/false, /*reuse_b=*/false);
}

// Load a 16x32 bf16 "A-pattern" fragment from a row-major matrix
// (also valid for B: B^T rows loaded A-style give the B operand layout).
// lane L in [0,16): row = L, K chunks {0..7, 16..23}
// lane L in [16,32): row = L-16, K chunks {8..15, 24..31}
__device__ __forceinline__ v16bf ld_frag(const __bf16* __restrict__ base, int ld) {
  const int lane = threadIdx.x & 31;
  const __bf16* p = base + (size_t)(lane & 15) * (size_t)ld + ((lane >> 4) << 3);
  union { v16bf v; uint4 u[2]; } f;
  f.u[0] = *(const uint4*)(p);
  f.u[1] = *(const uint4*)(p + 16);
  return f.v;
}

// Async global->LDS copy of one 16-byte chunk per lane (ASYNCcnt-tracked).
// LDS byte offset = low 32 bits of the generic shared-memory address
// (ISA: LDS aperture addresses map via ADDR[31:0]).
__device__ __forceinline__ void async_cp16(__bf16* lds_dst, const __bf16* gsrc) {
  uint32_t loff = (uint32_t)(uintptr_t)lds_dst;
  asm volatile("global_load_async_to_lds_b128 %0, %1, off"
               :: "v"(loff), "v"(gsrc)
               : "memory");
}

__device__ __forceinline__ void wait_async0() {
  asm volatile("s_wait_asynccnt 0x0" ::: "memory");
}

// ---------------- kernel 1: f32 -> bf16 weight conversion ----------------
__global__ void __launch_bounds__(256) cvt_bf16_kernel(const float* __restrict__ in,
                                                       __bf16* __restrict__ out, int n) {
  int i = blockIdx.x * 256 + threadIdx.x;
  if (i < n) out[i] = (__bf16)in[i];
}

// ---------------- kernel 2: x [b,C,N] f32 -> xt [b,N,C] bf16 ----------------
__global__ void __launch_bounds__(256) xpose_kernel(const float* __restrict__ x,
                                                    __bf16* __restrict__ xt) {
  __shared__ __bf16 tile[256 * 40]; // rows n (256), 32 c + 8 pad
  const int nb = N / 256;           // 25
  const int cb = C / 32;            // 16
  int bid  = blockIdx.x;
  int bidx = bid / (nb * cb);
  int r    = bid % (nb * cb);
  int c0   = (r / nb) * 32;
  int n0   = (r % nb) * 256;
  int t    = threadIdx.x;

  #pragma unroll
  for (int i = 0; i < 32; ++i) { // coalesced f32 reads along N
    float v = x[(size_t)(bidx * C + c0 + i) * N + n0 + t];
    tile[t * 40 + i] = (__bf16)v;
  }
  __syncthreads();
  __bf16* dst = xt + (size_t)(bidx * N + n0 + t) * C + c0;
  const uint4* src = (const uint4*)(tile + t * 40);
  uint4* d4 = (uint4*)dst;
  #pragma unroll
  for (int k = 0; k < 4; ++k) d4[k] = src[k];
}

// ---------------- kernel 3: projections ----------------
// kq[b,N,CK] = relu(BN(xt.wk^T + bk)),  vcn[b,CV,N] = xt.wv^T + bv
__global__ void __launch_bounds__(128) proj_kernel(
    const __bf16* __restrict__ xt, const __bf16* __restrict__ wkb,
    const __bf16* __restrict__ wvb, const float* __restrict__ bk,
    const float* __restrict__ gamma, const float* __restrict__ beta,
    const float* __restrict__ rmean, const float* __restrict__ rvar,
    const float* __restrict__ bv, __bf16* __restrict__ kq,
    __bf16* __restrict__ vcn) {
  const int wave = threadIdx.x >> 5;
  const int lane = threadIdx.x & 31;
  const int half = lane >> 4;
  const int lx   = lane & 15;
  const int tilesPerB = N / 16;
  const int bidx = blockIdx.x / tilesPerB;
  const int n0   = (blockIdx.x % tilesPerB) * 16;

  const __bf16* xrow = xt + (size_t)(bidx * N + n0) * C;
  v16bf xf[16]; // full 16 x 512 pixel tile, reused across all output tiles
  #pragma unroll
  for (int k = 0; k < 16; ++k) xf[k] = ld_frag(xrow + k * 32, C);

  if (wave < 2) {
    #pragma unroll 1
    for (int ti = 0; ti < 8; ++ti) {
      const int o0 = (wave * 8 + ti) * 16;
      v8f acc = v8zero();
      #pragma unroll
      for (int k = 0; k < 16; ++k) {
        v16bf bw = ld_frag(wkb + (size_t)o0 * C + k * 32, C); // B^T = wk rows
        acc = wmma_bf16(xf[k], bw, acc);                       // D[n][o]
      }
      const int o = o0 + lx;
      const float s  = gamma[o] * rsqrtf(rvar[o] + 1e-5f);
      const float tb = (bk[o] - rmean[o]) * s + beta[o];
      __bf16* dst = kq + ((size_t)(bidx * N + n0)) * CK + o;
      #pragma unroll
      for (int g = 0; g < 8; ++g) {
        float v = acc[g] * s + tb;
        v = v > 0.f ? v : 0.f;
        dst[(size_t)(g + half * 8) * CK] = (__bf16)v;
      }
    }
  } else {
    #pragma unroll 1
    for (int ti = 0; ti < 8; ++ti) {
      const int o0 = ((wave - 2) * 8 + ti) * 16;
      v8f acc = v8zero();
      #pragma unroll
      for (int k = 0; k < 16; ++k) {
        v16bf aw = ld_frag(wvb + (size_t)o0 * C + k * 32, C); // A = wv rows
        acc = wmma_bf16(aw, xf[k], acc);                       // D[o][n]
      }
      #pragma unroll
      for (int g = 0; g < 8; ++g) {
        const int o = o0 + g + half * 8;
        float v = acc[g] + bv[o];
        vcn[((size_t)bidx * CV + o) * N + n0 + lx] = (__bf16)v;
      }
    }
  }
}

// ---------------- kernel 4: flash attention + fused output projection ----------
// 128 threads = 4 waves; each wave owns 16 queries; K/V tiles (32 keys) are
// cooperatively double-buffered in LDS via async global->LDS loads.
__global__ void __launch_bounds__(128) flash_kernel(
    const __bf16* __restrict__ kq,   // [B,N,CK]  (Q and K share this)
    const __bf16* __restrict__ vcn,  // [B,CV,N]
    const __bf16* __restrict__ wWb,  // [CO,CV]
    const float* __restrict__ bW,    // [CO]
    float* __restrict__ out) {       // [B,CO,N]
  extern __shared__ __bf16 dsm[];
  const int wave = threadIdx.x >> 5;
  const int lane = threadIdx.x & 31;
  const int half = lane >> 4;
  const int lx   = lane & 15;
  const int tid  = threadIdx.x;
  const int tilesPerB = N / 64;
  const int bidx = blockIdx.x / tilesPerB;
  const int n0   = (blockIdx.x % tilesPerB) * 64 + wave * 16;

  const __bf16* kqb = kq + (size_t)bidx * N * CK;
  const __bf16* vb  = vcn + (size_t)bidx * CV * N;
  __bf16* pw = dsm + SM_PW0 + wave * P_ELEMS;

  // stage one 32-key tile of K (32x256) and V (256x32) into LDS buffer `buf`
  auto stage = [&](int buf, int m0) {
    __bf16* kb = dsm + buf * KT_ELEMS;
    #pragma unroll
    for (int i = 0; i < 8; ++i) {                 // 1024 chunks / 128 threads
      int j = i * 128 + tid;
      int row = j >> 5, ch = j & 31;              // 32 rows x 32 chunks(16B)
      async_cp16(kb + row * KT_STRIDE + ch * 8,
                 kqb + (size_t)(m0 + row) * CK + ch * 8);
    }
    __bf16* vt = dsm + SM_VT0 + buf * VT_ELEMS;
    #pragma unroll
    for (int i = 0; i < 8; ++i) {                 // 1024 chunks / 128 threads
      int j = i * 128 + tid;
      int row = j >> 2, ch = j & 3;               // 256 rows x 4 chunks(16B)
      async_cp16(vt + row * VT_STRIDE + ch * 8,
                 vb + (size_t)row * N + m0 + ch * 8);
    }
  };

  // cache the 16x256 Q tile as 8 A-fragments (L2-resident, one-time)
  v16bf qf[8];
  #pragma unroll
  for (int k = 0; k < 8; ++k)
    qf[k] = ld_frag(kqb + (size_t)n0 * CK + k * 32, CK);

  v8f oacc[16];
  #pragma unroll
  for (int t = 0; t < 16; ++t) oacc[t] = v8zero();
  float mrow[8], lrow[8];
  #pragma unroll
  for (int g = 0; g < 8; ++g) { mrow[g] = -1e30f; lrow[g] = 0.f; }

  stage(0, 0);
  int cur = 0;
  #pragma unroll 1
  for (int m0 = 0; m0 < N; m0 += 32) {
    wait_async0();        // this wave's staged chunks have landed in LDS
    __syncthreads();      // ... and everyone else's too
    if (m0 + 32 < N) stage(cur ^ 1, m0 + 32);  // prefetch next tile

    const __bf16* kb = dsm + cur * KT_ELEMS;
    const __bf16* vt = dsm + SM_VT0 + cur * VT_ELEMS;

    // S = Q K^T for 16 queries x 32 keys (two 16x16 D tiles)
    v8f s0 = v8zero(), s1 = v8zero();
    #pragma unroll
    for (int k = 0; k < 8; ++k) {
      v16bf b0 = ld_frag(kb + k * 32, KT_STRIDE);
      v16bf b1 = ld_frag(kb + 16 * KT_STRIDE + k * 32, KT_STRIDE);
      s0 = wmma_bf16(qf[k], b0, s0);
      s1 = wmma_bf16(qf[k], b1, s1);
    }
    // online softmax; row of D = (g + 8*half), cols across 16 lanes of half
    float alpha[8];
    #pragma unroll
    for (int g = 0; g < 8; ++g) {
      float a  = s0[g] * 0.0625f;  // Ck^-0.5 = 1/16
      float b2 = s1[g] * 0.0625f;
      float mx = fmaxf(a, b2);
      #pragma unroll
      for (int d = 1; d < 16; d <<= 1) mx = fmaxf(mx, __shfl_xor(mx, d, 32));
      const float mnew = fmaxf(mrow[g], mx);
      const float p0 = __expf(a - mnew);
      const float p1 = __expf(b2 - mnew);
      alpha[g] = __expf(mrow[g] - mnew);
      mrow[g] = mnew;
      float rs = p0 + p1;
      #pragma unroll
      for (int d = 1; d < 16; d <<= 1) rs += __shfl_xor(rs, d, 32);
      lrow[g] = lrow[g] * alpha[g] + rs;
      const int q = g + half * 8;
      pw[q * P_STRIDE + lx]      = (__bf16)p0;
      pw[q * P_STRIDE + 16 + lx] = (__bf16)p1;
    }
    #pragma unroll
    for (int t = 0; t < 16; ++t)
      #pragma unroll
      for (int g = 0; g < 8; ++g) oacc[t][g] *= alpha[g];
    // P back out of LDS in A-fragment layout (per-wave region, in-wave ds order)
    v16bf pA = ld_frag(pw, P_STRIDE);
    #pragma unroll
    for (int t = 0; t < 16; ++t) {
      v16bf bv2 = ld_frag(vt + (size_t)(t * 16) * VT_STRIDE, VT_STRIDE);
      oacc[t] = wmma_bf16(pA, bv2, oacc[t]);     // D[q][cv]
    }
    __syncthreads();      // all waves done with buf `cur`; safe to restage
    cur ^= 1;
  }

  // normalize and stage O (16 x 256 bf16) in LDS (reuses K-buffer region)
  __bf16* ow = dsm + wave * (16 * O_STRIDE);
  #pragma unroll
  for (int g = 0; g < 8; ++g) {
    const float inv = 1.f / lrow[g];
    const int q = g + half * 8;
    #pragma unroll
    for (int t = 0; t < 16; ++t)
      ow[q * O_STRIDE + t * 16 + lx] = (__bf16)(oacc[t][g] * inv);
  }

  // fused output projection: out[co][nq] = wW[co,:] . O[nq,:] + bW[co]
  float* outb = out + (size_t)bidx * CO * N;
  #pragma unroll 1
  for (int ct = 0; ct < CO / 16; ++ct) {
    v8f acc = v8zero();
    #pragma unroll
    for (int k = 0; k < 8; ++k) {
      v16bf aw = ld_frag(wWb + (size_t)(ct * 16) * CV + k * 32, CV); // A = wW rows
      v16bf bo = ld_frag(ow + k * 32, O_STRIDE);                     // B^T = O rows
      acc = wmma_bf16(aw, bo, acc);                                  // D[co][nq]
    }
    #pragma unroll
    for (int g = 0; g < 8; ++g) {
      const int co = ct * 16 + g + half * 8;
      outb[(size_t)co * N + n0 + lx] = acc[g] + bW[co]; // coalesced f32 store
    }
  }
}

// ---------------- host launcher ----------------
extern "C" void kernel_launch(void* const* d_in, const int* in_sizes, int n_in,
                              void* d_out, int out_size, void* d_ws, size_t ws_size,
                              hipStream_t stream) {
  const float* x     = (const float*)d_in[0];
  const float* wk    = (const float*)d_in[1];
  const float* bk    = (const float*)d_in[2];
  const float* gamma = (const float*)d_in[3];
  const float* beta  = (const float*)d_in[4];
  const float* rmean = (const float*)d_in[5];
  const float* rvar  = (const float*)d_in[6];
  const float* wv    = (const float*)d_in[7];
  const float* bv    = (const float*)d_in[8];
  const float* wW    = (const float*)d_in[9];
  const float* bW    = (const float*)d_in[10];
  float* out = (float*)d_out;

  // workspace carve-up (bf16 buffers), 256B aligned
  char* ws = (char*)d_ws;
  auto align256 = [](size_t v) { return (v + 255) & ~(size_t)255; };
  size_t off = 0;
  __bf16* xt  = (__bf16*)(ws + off); off = align256(off + (size_t)Bn * N * C  * 2);
  __bf16* kq  = (__bf16*)(ws + off); off = align256(off + (size_t)Bn * N * CK * 2);
  __bf16* vcn = (__bf16*)(ws + off); off = align256(off + (size_t)Bn * CV * N * 2);
  __bf16* wkb = (__bf16*)(ws + off); off = align256(off + (size_t)CK * C * 2);
  __bf16* wvb = (__bf16*)(ws + off); off = align256(off + (size_t)CV * C * 2);
  __bf16* wWb = (__bf16*)(ws + off); off = align256(off + (size_t)CO * CV * 2);
  (void)ws_size; (void)n_in; (void)in_sizes; (void)out_size;

  // 1) weights -> bf16
  cvt_bf16_kernel<<<(CK * C + 255) / 256, 256, 0, stream>>>(wk, wkb, CK * C);
  cvt_bf16_kernel<<<(CV * C + 255) / 256, 256, 0, stream>>>(wv, wvb, CV * C);
  cvt_bf16_kernel<<<(CO * CV + 255) / 256, 256, 0, stream>>>(wW, wWb, CO * CV);

  // 2) x -> xt (transpose + bf16)
  xpose_kernel<<<Bn * (C / 32) * (N / 256), 256, 0, stream>>>(x, xt);

  // 3) K/Q and V projections (BN+ReLU folded for K)
  proj_kernel<<<Bn * (N / 16), 128, 0, stream>>>(xt, wkb, wvb, bk, gamma, beta,
                                                 rmean, rvar, bv, kq, vcn);

  // 4) flash attention + fused output projection (dynamic LDS, double-buffered
  //    async K/V staging)
  flash_kernel<<<Bn * (N / 64), 128, SM_BYTES, stream>>>(kq, vcn, wWb, bW, out);
}
